// DocSelfAttention_23098334118563
// MI455X (gfx1250) — matrix operand order repaired and checked by reading, not compiled, not run on hardware
//
#include <hip/hip_runtime.h>
#include <hip/hip_bf16.h>

// B=8, A=512, E=128, H=512, M=256
// workspace layout (floats):
//   P    : 4096*256   (word_all @ w1 + b1)          4 MB
//   Q    : 1024*256   (word_weighted @ w1)          1 MB
//   attn : 8*128*512                                 2 MB
//   out2 : 1024*512   (pooled + word_weighted)       2 MB

typedef __attribute__((ext_vector_type(16))) __bf16 v16bf;
typedef __attribute__((ext_vector_type(8)))  float  v8f;

// Branch-free fast tanh on the hardware TRANS pipe:
//   tanh(x) = 1 - 2/(exp(2x)+1);  exp->+inf => 1, exp->0 => -1. No NaN paths.
__device__ __forceinline__ float tanh_fast(float x) {
  const float e = __expf(2.0f * x);               // v_exp_f32
  return 1.0f - 2.0f * __builtin_amdgcn_rcpf(e + 1.0f);  // v_rcp_f32 + fma
}

// Row-major GEMM: C[row,n] = sum_k Af[row,k]*Bf[k,n] (+bias[n]) (+addmat[row,n])
// bf16 WMMA 16x16x32, f32 accumulate. Block = 256 thr = 8 waves -> 16x128 strip.
// K,N compile-time so strided B gathers use immediate offsets (clause-able).
// A-chunk (16x64) staged in LDS once per block (shared by all 8 waves).
// grid.x = N/128, grid.y = rows/16, grid.z = batch.
template <int K, int N, bool HAS_BIAS, bool HAS_ADD>
__global__ __launch_bounds__(256) void wmma_gemm_bf16(
    const float* __restrict__ Af, const float* __restrict__ Bf,
    float* __restrict__ C, const float* __restrict__ bias,
    const float* __restrict__ addmat, long sA, long sB, long sC) {
  __shared__ float As[16][68];    // 64 K-values + 4 pad (b128 alignment kept)
  __shared__ float Cs[16][132];   // 128 cols + 4 pad

  const int b = blockIdx.z;
  Af += (long)b * sA;
  Bf += (long)b * sB;
  C  += (long)b * sC;
  const float* addp = HAS_ADD ? (addmat + (long)b * sC) : nullptr;

  const int tid  = threadIdx.x;
  const int lane = tid & 31;
  const int wave = tid >> 5;
  const int row0 = blockIdx.y * 16;
  const int colB = blockIdx.x * 128;
  const int col0 = colB + wave * 16;

  // cooperative A-stage mapping: 16 rows x 64 K, one float4 per thread
  const int arow = tid >> 4;
  const int acol = (tid & 15) * 4;
  const float* aload = Af + (long)(row0 + arow) * K + acol;

  // fragment addressing (contiguous k = 16*(lane/16) + e, same for A and B)
  const int frow = lane & 15;                 // A row within tile
  const int akb  = (lane < 16) ? 0 : 16;      // A K-half select
  const int bcol = col0 + (lane & 15);        // B column
  const int bkb  = (lane < 16) ? 0 : 16;      // B K-half select
  const float* bbase = Bf + (long)bkb * N + bcol;

  v8f acc = {};
  for (int k0 = 0; k0 < K; k0 += 64) {
    __syncthreads();  // previous chunk's LDS reads done
    *(float4*)&As[arow][acol] = *(const float4*)(aload + k0);
    __builtin_prefetch(aload + k0 + 64, 0, 3);
    __builtin_prefetch(bbase + (long)(k0 + 64) * N, 0, 3);
    __syncthreads();

#pragma unroll
    for (int half = 0; half < 2; ++half) {
      const int kh = half * 32;

      // A fragment: 4x ds_load_b128 from the shared chunk
      float a16[16];
      *(float4*)(a16 + 0)  = *(const float4*)&As[frow][kh + akb + 0];
      *(float4*)(a16 + 4)  = *(const float4*)&As[frow][kh + akb + 4];
      *(float4*)(a16 + 8)  = *(const float4*)&As[frow][kh + akb + 8];
      *(float4*)(a16 + 12) = *(const float4*)&As[frow][kh + akb + 12];
      v16bf afrag;
#pragma unroll
      for (int e = 0; e < 16; ++e) afrag[e] = (__bf16)a16[e];

      // B fragment: 16 loads, compile-time immediate offsets (one clause)
      const float* bp = bbase + (long)(k0 + kh) * N;
      v16bf bfrag;
#pragma unroll
      for (int e = 0; e < 16; ++e) bfrag[e] = (__bf16)bp[e * N];

      acc = __builtin_amdgcn_wmma_f32_16x16x32_bf16(
          false, afrag, false, bfrag, (short)0, acc, false, false);
    }
  }

  // spill accumulator tile to LDS (wmma C/D layout), then coalesced b128 stores
  __syncthreads();
  {
    const int crow = (lane < 16) ? 0 : 8;
    const int ccol = wave * 16 + (lane & 15);
#pragma unroll
    for (int r = 0; r < 8; ++r) Cs[crow + r][ccol] = acc[r];
  }
  __syncthreads();
  {
    const int row = tid >> 4;          // 0..15
    const int col = (tid & 15) * 8;    // 0..120
    float4 v0 = *(const float4*)&Cs[row][col];
    float4 v1 = *(const float4*)&Cs[row][col + 4];
    const long gbase = (long)(row0 + row) * N + colB + col;
    if (HAS_BIAS) {
      const float4 b0  = *(const float4*)(bias + colB + col);
      const float4 b1v = *(const float4*)(bias + colB + col + 4);
      v0.x += b0.x;  v0.y += b0.y;  v0.z += b0.z;  v0.w += b0.w;
      v1.x += b1v.x; v1.y += b1v.y; v1.z += b1v.z; v1.w += b1v.w;
    }
    if (HAS_ADD) {
      const float4 a0 = *(const float4*)(addp + gbase);
      const float4 a1 = *(const float4*)(addp + gbase + 4);
      v0.x += a0.x; v0.y += a0.y; v0.z += a0.z; v0.w += a0.w;
      v1.x += a1.x; v1.y += a1.y; v1.z += a1.z; v1.w += a1.w;
    }
    *(float4*)(C + gbase)     = v0;
    *(float4*)(C + gbase + 4) = v1;
  }
}

// scores[b,e,a] = sum_m tanh(P[b,a,m] - Q[b,e,m]) * w2[m] + b2 ; softmax over a.
// One block (256 threads) per (b,e) pair; A=512, M=256. (b1 folded into P.)
__global__ __launch_bounds__(256) void scores_softmax(
    const float* __restrict__ P,   // [B*A, 256]
    const float* __restrict__ Q,   // [B*E, 256]
    const float* __restrict__ w2, const float* __restrict__ b2,
    float* __restrict__ attn) {    // [B*E, 512]
  const int ADIM = 512, MDIM = 256;
  const int be  = blockIdx.x;      // b*128 + e
  const int b   = be >> 7;
  const int tid = threadIdx.x;

  __shared__ float qv[256];
  __shared__ float wv[256];
  __shared__ float sc[512];
  __shared__ float red[256];

  qv[tid] = Q[(long)be * MDIM + tid];
  wv[tid] = w2[tid];
  __syncthreads();

  const float b2v = b2[0];
  for (int a = tid; a < ADIM; a += 256) {
    const float* prow = P + ((long)b * ADIM + a) * MDIM;
    float s = 0.0f;
#pragma unroll 2
    for (int m = 0; m < MDIM; m += 4) {
      const float4 p = *(const float4*)(prow + m);
      s += tanh_fast(p.x - qv[m + 0]) * wv[m + 0];
      s += tanh_fast(p.y - qv[m + 1]) * wv[m + 1];
      s += tanh_fast(p.z - qv[m + 2]) * wv[m + 2];
      s += tanh_fast(p.w - qv[m + 3]) * wv[m + 3];
    }
    sc[a] = s + b2v;
  }
  __syncthreads();

  red[tid] = fmaxf(sc[tid], sc[tid + 256]);
  __syncthreads();
  for (int s = 128; s > 0; s >>= 1) {
    if (tid < s) red[tid] = fmaxf(red[tid], red[tid + s]);
    __syncthreads();
  }
  const float mx = red[0];
  __syncthreads();

  const float e0 = __expf(sc[tid] - mx);
  const float e1 = __expf(sc[tid + 256] - mx);
  red[tid] = e0 + e1;
  __syncthreads();
  for (int s = 128; s > 0; s >>= 1) {
    if (tid < s) red[tid] += red[tid + s];
    __syncthreads();
  }
  const float inv = __builtin_amdgcn_rcpf(red[0]);

  attn[(long)be * ADIM + tid]       = e0 * inv;
  attn[(long)be * ADIM + tid + 256] = e1 * inv;
}

extern "C" void kernel_launch(void* const* d_in, const int* in_sizes, int n_in,
                              void* d_out, int out_size, void* d_ws, size_t ws_size,
                              hipStream_t stream) {
  (void)in_sizes; (void)n_in; (void)out_size; (void)ws_size;
  const float* word_all      = (const float*)d_in[0];  // [8,512,512]
  const float* word_weighted = (const float*)d_in[1];  // [8,128,512]
  const float* w1 = (const float*)d_in[2];             // [512,256]
  const float* b1 = (const float*)d_in[3];             // [256]
  const float* w2 = (const float*)d_in[4];             // [256]
  const float* b2 = (const float*)d_in[5];             // scalar
  const float* w3 = (const float*)d_in[6];             // [512,256]
  const float* b3 = (const float*)d_in[7];             // [256]
  float* out = (float*)d_out;                          // [8,128,256]

  float* P    = (float*)d_ws;            // 4096*256
  float* Q    = P    + 4096 * 256;       // 1024*256
  float* attn = Q    + 1024 * 256;       // 8*128*512
  float* out2 = attn + 8 * 128 * 512;    // 1024*512

  const dim3 blk(256);

  // P = word_all(4096x512) @ w1(512x256) + b1
  wmma_gemm_bf16<512, 256, true, false><<<dim3(2, 256, 1), blk, 0, stream>>>(
      word_all, w1, P, b1, nullptr, 0, 0, 0);

  // Q = word_weighted(1024x512) @ w1(512x256)
  wmma_gemm_bf16<512, 256, false, false><<<dim3(2, 64, 1), blk, 0, stream>>>(
      word_weighted, w1, Q, nullptr, nullptr, 0, 0, 0);

  // scores + softmax over A
  scores_softmax<<<dim3(1024), blk, 0, stream>>>(P, Q, w2, b2, attn);

  // pooled[b] = attn[b](128x512) @ word_all[b](512x512) + word_weighted[b]
  wmma_gemm_bf16<512, 512, false, true><<<dim3(4, 8, 8), blk, 0, stream>>>(
      attn, word_all, out2, nullptr, word_weighted,
      128L * 512, 512L * 512, 128L * 512);

  // out = out2(1024x512) @ w3(512x256) + b3
  wmma_gemm_bf16<512, 256, true, false><<<dim3(2, 64, 1), blk, 0, stream>>>(
      out2, w3, out, b3, nullptr, 0, 0, 0);
}
